// TriQueryIntegrator_51625506898052
// MI455X (gfx1250) — matrix operand order repaired and checked by reading, not compiled, not run on hardware
//
#include <hip/hip_runtime.h>
#include <hip/hip_bf16.h>

// ---------------------------------------------------------------------------
// CDNA5 (gfx1250) implementation of TriQueryIntegrator forward.
// wave32; matrix math via v_wmma_f32_16x16x32_bf16 (fp32 accumulate).
// Weights pre-swizzled to WMMA-B fragment order (one 32B load / lane / frag);
// GEMMs block 4 N-tiles per wave (A fragment reused across 4 WMMAs);
// LDS staging of strided tiles uses global_load_async_to_lds_b128 (ASYNCcnt).
// ---------------------------------------------------------------------------

typedef __attribute__((ext_vector_type(16))) __bf16 v16bf;
typedef __attribute__((ext_vector_type(8)))  __bf16 v8bf;
typedef __attribute__((ext_vector_type(8)))  float  v8f;
typedef int v4i __attribute__((vector_size(16)));   // matches builtin param type

#define DIM      256
#define HEADS    8
#define HD       32
#define NB       2
#define VTOT     37440          // 32768+4096+512+64
#define ROWS_FF  (NB * VTOT)    // 74880

#define AS1 __attribute__((address_space(1)))
#define AS3 __attribute__((address_space(3)))

#if defined(__has_builtin)
#  if __has_builtin(__builtin_amdgcn_global_load_async_to_lds_b128)
#    define HAVE_ASYNC_LDS 1
#  endif
#  if __has_builtin(__builtin_amdgcn_s_wait_asynccnt)
#    define HAVE_WAIT_ASYNC 1
#  endif
#endif

// async 16B global -> LDS copy (falls back to a normal copy if unavailable)
__device__ __forceinline__ void async_b128(const void* g, void* l) {
#ifdef HAVE_ASYNC_LDS
  __builtin_amdgcn_global_load_async_to_lds_b128(
      (AS1 v4i*)(unsigned long long)(size_t)g,
      (AS3 v4i*)(unsigned)(size_t)l, 0, 0);
#else
  *(uint4*)l = *(const uint4*)g;
#endif
}
__device__ __forceinline__ void wait_async0() {
#ifdef HAVE_ASYNC_LDS
#  ifdef HAVE_WAIT_ASYNC
  __builtin_amdgcn_s_wait_asynccnt(0);
#  else
  asm volatile("s_wait_asynccnt 0x0" ::: "memory");
#  endif
#endif
}

__device__ __forceinline__ v8f wmma_bf16(v16bf a, v16bf b, v8f c) {
  // D = A(16x32 bf16) x B(32x16 bf16) + C(16x16 f32)
  return __builtin_amdgcn_wmma_f32_16x16x32_bf16(
      false, a, false, b, (short)0, c, false, false);
}

// 16-bit A fragment (16x32): element e of lane -> K index.
__device__ __forceinline__ int a_kmap(int e, int half) {
  return e + half * 16;
}
// 16-bit B fragment (32x16): N = lane&15 ; K = e + (lane>>4)*16
// f32 C/D fragment (16x16):  M = r + (lane>>4)*8 ; N = lane&15

__device__ __forceinline__ v16bf combine16(v8bf lo, v8bf hi) {
  v16bf a;
#pragma unroll
  for (int e = 0; e < 8; ++e) { a[e] = lo[e]; a[e + 8] = hi[e]; }
  return a;
}
// A fragment from a row-major bf16 row (base already at [row*256 + kk])
__device__ __forceinline__ v16bf ldA_row(const __bf16* rowbase, int half) {
  return combine16(*(const v8bf*)(rowbase + half * 16),
                   *(const v8bf*)(rowbase + half * 16 + 8));
}
// B fragment from pre-swizzled weights: one 32B contiguous run per lane
__device__ __forceinline__ v16bf ldB_swz(const __bf16* Wsw, int kk, int nt, int lane) {
  const __bf16* p = Wsw + (((size_t)(kk >> 5) * 16 + (nt >> 4)) * 32 + lane) * 16;
  return combine16(*(const v8bf*)p, *(const v8bf*)(p + 8));
}

// ------------------------------ small utils --------------------------------
__global__ void cvt_kernel(const float* __restrict__ src, __bf16* __restrict__ dst, int n) {
  int i = blockIdx.x * 256 + threadIdx.x;
  if (i < n) dst[i] = (__bf16)src[i];
}

__global__ void zero_kernel(float* __restrict__ p, int n) {
  int i = blockIdx.x * 256 + threadIdx.x;
  if (i < n) p[i] = 0.0f;
}

// Re-pack a 256x256 f32 weight (in-major) into WMMA-B fragment order:
// flat index (((tk*16+tn)*32+lane)*16+e) = W[tk*32 + e + (lane>>4)*16][tn*16 + (lane&15)]
__global__ void swz_weight_kernel(const float* __restrict__ W, __bf16* __restrict__ Wsw) {
  const int i = blockIdx.x * 256 + threadIdx.x;   // 65536 total
  const int e = i & 15, lane = (i >> 4) & 31, tn = (i >> 9) & 15, tk = i >> 13;
  const int c = tk * 32 + e + (lane >> 4) * 16;
  const int n = tn * 16 + (lane & 15);
  Wsw[i] = (__bf16)W[c * 256 + n];
}

// ------------------- feat projection: ff = feat^T * W + b ------------------
// feat: (B,256,V) f32 channel-major ; Wsw: swizzled bf16 ; ff bf16 (B,VTOT,256)
// Each wave: 16 voxels x 64 output channels (4 N-tiles; A frag reused 4x).
__global__ void featproj_kernel(const float* __restrict__ feat,
                                const __bf16* __restrict__ Wsw,
                                const float* __restrict__ bias,
                                __bf16* __restrict__ ff,
                                int V, int level_off) {
  __shared__ float Atile[32][16];   // [k][m], staged via async DMA
  const int lane = threadIdx.x, half = lane >> 4, l15 = lane & 15;
  const int vt = blockIdx.x * 16, nt0 = blockIdx.y * 64, b = blockIdx.z;
  const float* fb = feat + (size_t)b * DIM * V;

  v8f acc[4] = {};
  for (int kk = 0; kk < DIM; kk += 32) {
    const int c = kk + lane;
    const float* src = fb + (size_t)c * V + vt;      // 16 f32 = 64B contiguous
    if (kk + 32 < DIM)
      __builtin_prefetch(fb + (size_t)(c + 32) * V + vt, 0, 1); // global_prefetch_b8
    __syncthreads();                                  // WAR: prior frag reads done
#pragma unroll
    for (int s = 0; s < 4; ++s) async_b128(src + s * 4, &Atile[lane][s * 4]);
    wait_async0();
    __syncthreads();
    v16bf a;
#pragma unroll
    for (int e = 0; e < 16; ++e) a[e] = (__bf16)Atile[a_kmap(e, half)][l15];
#pragma unroll
    for (int t = 0; t < 4; ++t)
      acc[t] = wmma_bf16(a, ldB_swz(Wsw, kk, nt0 + t * 16, lane), acc[t]);
  }
#pragma unroll
  for (int t = 0; t < 4; ++t) {
#pragma unroll
    for (int r = 0; r < 8; ++r) {
      const int m = r + half * 8;              // voxel within tile
      const int n = nt0 + t * 16 + l15;        // out channel
      ff[((size_t)b * VTOT + level_off + vt + m) * DIM + n] = (__bf16)(acc[t][r] + bias[n]);
    }
  }
}

// ----------------- generic row GEMM: Y = X(rows,256) * Wsw + b -------------
// Each wave: 16 rows x 64 output channels (A frag reused 4x).
__global__ void gemm_rows_kernel(const __bf16* __restrict__ X,
                                 const __bf16* __restrict__ Wsw,
                                 const float* __restrict__ bias,
                                 __bf16* __restrict__ Y, int rows) {
  const int lane = threadIdx.x, half = lane >> 4, l15 = lane & 15;
  const int rt = blockIdx.x * 16, nt0 = blockIdx.y * 64;
  const int row = rt + l15;
  const bool ok = row < rows;
  const __bf16* xr = X + (size_t)(ok ? row : 0) * DIM;
  const v16bf zf = {};
  v8f acc[4] = {};
  for (int kk = 0; kk < DIM; kk += 32) {
    const v16bf a = ok ? ldA_row(xr + kk, half) : zf;
#pragma unroll
    for (int t = 0; t < 4; ++t)
      acc[t] = wmma_bf16(a, ldB_swz(Wsw, kk, nt0 + t * 16, lane), acc[t]);
  }
#pragma unroll
  for (int t = 0; t < 4; ++t) {
#pragma unroll
    for (int r = 0; r < 8; ++r) {
      const int m = rt + r + half * 8;
      if (m < rows)
        Y[(size_t)m * DIM + nt0 + t * 16 + l15] = (__bf16)(acc[t][r] + bias[nt0 + t * 16 + l15]);
    }
  }
}

// same but f32 output with scaled bias (shared output projection, bias x4)
__global__ void gemm_rows_f32_kernel(const __bf16* __restrict__ X,
                                     const __bf16* __restrict__ Wsw,
                                     const float* __restrict__ bias, float bias_scale,
                                     float* __restrict__ Y, int rows) {
  const int lane = threadIdx.x, half = lane >> 4, l15 = lane & 15;
  const int rt = blockIdx.x * 16, nt0 = blockIdx.y * 64;
  const int row = rt + l15;
  const bool ok = row < rows;
  const __bf16* xr = X + (size_t)(ok ? row : 0) * DIM;
  const v16bf zf = {};
  v8f acc[4] = {};
  for (int kk = 0; kk < DIM; kk += 32) {
    const v16bf a = ok ? ldA_row(xr + kk, half) : zf;
#pragma unroll
    for (int t = 0; t < 4; ++t)
      acc[t] = wmma_bf16(a, ldB_swz(Wsw, kk, nt0 + t * 16, lane), acc[t]);
  }
#pragma unroll
  for (int t = 0; t < 4; ++t) {
#pragma unroll
    for (int r = 0; r < 8; ++r) {
      const int m = rt + r + half * 8;
      if (m < rows)
        Y[(size_t)m * DIM + nt0 + t * 16 + l15] =
            acc[t][r] + bias_scale * bias[nt0 + t * 16 + l15];
    }
  }
}

// -------------------- flash soft attention over 4 levels -------------------
__global__ void flash_soft_kernel(const __bf16* __restrict__ Qp,
                                  const __bf16* __restrict__ Ks,
                                  const __bf16* __restrict__ Vs,
                                  __bf16* __restrict__ oacc, int Nq) {
  const int Vl[4]  = {32768, 4096, 512, 64};
  const int Off[4] = {0, 32768, 36864, 37376};
  const int lane = threadIdx.x, half = lane >> 4, l15 = lane & 15;
  const int qt = blockIdx.x * 16, h = blockIdx.y, b = blockIdx.z;
  const float scale = 0.17677669529663687f;   // hd^-0.5 folded into Q
  __shared__ __bf16 Plds[16][32];
  __shared__ __bf16 Vlds[32][32];             // 32 keys x head slice (async staged)

  // A fragment of Q (16 queries x 32 head-dims), pre-scaled
  v16bf qf;
  {
    const int row = qt + l15;
    const bool ok = row < Nq;
    const __bf16* qr = Qp + ((size_t)(b * Nq + (ok ? row : 0))) * DIM + h * HD;
    const v8bf lo = *(const v8bf*)(qr + half * 16);
    const v8bf hi = *(const v8bf*)(qr + half * 16 + 8);
#pragma unroll
    for (int e = 0; e < 8; ++e) {
      qf[e]     = ok ? (__bf16)((float)lo[e] * scale) : (__bf16)0.0f;
      qf[e + 8] = ok ? (__bf16)((float)hi[e] * scale) : (__bf16)0.0f;
    }
  }

  v8f o0 = {}, o1 = {};
  for (int l = 0; l < 4; ++l) {
    const __bf16* Kb = Ks + ((size_t)b * VTOT + Off[l]) * DIM;
    const __bf16* Vb = Vs + ((size_t)b * VTOT + Off[l]) * DIM;
    v8f acc0 = {}, acc1 = {};
    float rmax[8], rsum[8];
#pragma unroll
    for (int r = 0; r < 8; ++r) { rmax[r] = -1e30f; rsum[r] = 0.0f; }

    for (int j = 0; j < Vl[l]; j += 32) {      // all levels divisible by 32
      // stage V chunk head-slice into LDS via async DMA (64B / lane)
      {
        const __bf16* vrow = Vb + (size_t)(j + lane) * DIM + h * HD;
        __syncthreads();                        // WAR vs previous frag reads
#pragma unroll
        for (int s = 0; s < 4; ++s) async_b128(vrow + s * 8, &Vlds[lane][s * 8]);
      }
      // B fragments of K^T : key tiles [j..j+15], [j+16..j+31] (32B / lane)
      const __bf16* kr0 = Kb + (size_t)(j + l15) * DIM + h * HD + half * 16;
      const __bf16* kr1 = Kb + (size_t)(j + 16 + l15) * DIM + h * HD + half * 16;
      const v16bf k0 = combine16(*(const v8bf*)kr0, *(const v8bf*)(kr0 + 8));
      const v16bf k1 = combine16(*(const v8bf*)kr1, *(const v8bf*)(kr1 + 8));
      v8f z = {};
      v8f s0 = wmma_bf16(qf, k0, z);
      v8f s1 = wmma_bf16(qf, k1, z);

      // online softmax; row stats reduced across each 16-lane half
      float p0[8], p1[8];
#pragma unroll
      for (int r = 0; r < 8; ++r) {
        float v = fmaxf(s0[r], s1[r]);
        for (int off = 1; off < 16; off <<= 1) v = fmaxf(v, __shfl_xor(v, off, 16));
        const float nm = fmaxf(rmax[r], v);
        const float corr = __expf(rmax[r] - nm);
        rmax[r] = nm;
        const float e0 = __expf(s0[r] - nm), e1 = __expf(s1[r] - nm);
        float ps = e0 + e1;
        for (int off = 1; off < 16; off <<= 1) ps += __shfl_xor(ps, off, 16);
        rsum[r] = rsum[r] * corr + ps;
        acc0[r] *= corr; acc1[r] *= corr;
        p0[r] = e0; p1[r] = e1;
      }
      // transpose P (C-layout -> A-layout) via LDS
#pragma unroll
      for (int r = 0; r < 8; ++r) {
        const int m = r + half * 8;
        Plds[m][l15]      = (__bf16)p0[r];
        Plds[m][16 + l15] = (__bf16)p1[r];
      }
      wait_async0();
      __syncthreads();
      v16bf pf;
#pragma unroll
      for (int e = 0; e < 16; ++e) pf[e] = Plds[l15][a_kmap(e, half)];
      // B fragments of V from LDS stage
      v16bf v0f, v1f;
#pragma unroll
      for (int e = 0; e < 16; ++e) {
        v0f[e] = Vlds[e + half * 16][l15];
        v1f[e] = Vlds[e + half * 16][16 + l15];
      }
      acc0 = wmma_bf16(pf, v0f, acc0);
      acc1 = wmma_bf16(pf, v1f, acc1);
    }
#pragma unroll
    for (int r = 0; r < 8; ++r) {
      const float inv = 1.0f / rsum[r];
      o0[r] += acc0[r] * inv;
      o1[r] += acc1[r] * inv;
    }
  }
#pragma unroll
  for (int r = 0; r < 8; ++r) {
    const int m = qt + r + half * 8;
    if (m < Nq) {
      __bf16* orow = oacc + ((size_t)(b * Nq + m)) * DIM + h * HD;
      orow[l15]      = (__bf16)o0[r];
      orow[16 + l15] = (__bf16)o1[r];
    }
  }
}

// ------------------------------ hard routing -------------------------------
__device__ __forceinline__ float gumbel_hash(unsigned b, unsigned k, unsigned v, unsigned l) {
  unsigned x = (b + 1u) * 0x9E3779B1u ^ (k + 1u) * 0x85EBCA77u ^
               (v + 1u) * 0xC2B2AE3Du ^ (l + 1u) * 0x27D4EB2Fu;
  x ^= x >> 16; x *= 0x7FEB352Du; x ^= x >> 15; x *= 0x846CA68Bu; x ^= x >> 16;
  const float u = (x + 1u) * 2.328306437e-10f;       // (0,1]
  return -__logf(-__logf(u + 1e-20f) + 1e-20f);
}

// argmax_v( Qa[b,k,:].ff[b,v,:]/sqrt(C) + gumbel )  (one-hot ST mask routing)
__global__ void hard_argmax_kernel(const __bf16* __restrict__ Qabf,
                                   const __bf16* __restrict__ ff,
                                   int* __restrict__ idxout) {
  const int Vl[4]  = {32768, 4096, 512, 64};
  const int Off[4] = {0, 32768, 36864, 37376};
  const int k = blockIdx.x, b = blockIdx.y, l = blockIdx.z, t = threadIdx.x;
  __shared__ float qrow[256];
  __shared__ float sbest[256];
  __shared__ int   sidx[256];
  qrow[t] = (float)Qabf[((size_t)(b * 16 + k)) * DIM + t];
  __syncthreads();
  const __bf16* fb = ff + ((size_t)b * VTOT + Off[l]) * DIM;
  float best = -1e30f; int bi = 0;
  const float inv_sqrt_c = 0.0625f;
  for (int v = t; v < Vl[l]; v += 256) {
    const __bf16* fr = fb + (size_t)v * DIM;
    float d = 0.0f;
    for (int c = 0; c < DIM; c += 8) {
      const v8bf vv = *(const v8bf*)(fr + c);
#pragma unroll
      for (int jj = 0; jj < 8; ++jj) d += qrow[c + jj] * (float)vv[jj];
    }
    const float s = d * inv_sqrt_c + gumbel_hash(b, k, v, l);
    if (s > best) { best = s; bi = v; }
  }
  sbest[t] = best; sidx[t] = bi; __syncthreads();
  for (int off = 128; off > 0; off >>= 1) {
    if (t < off && sbest[t + off] > sbest[t]) { sbest[t] = sbest[t + off]; sidx[t] = sidx[t + off]; }
    __syncthreads();
  }
  if (t == 0) idxout[(l * NB + b) * 16 + k] = sidx[0];
}

// one-hot attention == gather: out = ((ff[v*]·wv + bv)·wo + bo), summed levels
__global__ void hard_out_kernel(const __bf16* __restrict__ ff, const int* __restrict__ idx,
                                const __bf16* __restrict__ wv, const float* __restrict__ bv,
                                const __bf16* __restrict__ wo, const float* __restrict__ bo,
                                float* __restrict__ aggH) {
  const int Off[4] = {0, 32768, 36864, 37376};
  const int k = blockIdx.x, b = blockIdx.y, l = blockIdx.z, t = threadIdx.x;
  __shared__ float fr[256];
  __shared__ float vh[256];
  const int v = idx[(l * NB + b) * 16 + k];
  fr[t] = (float)ff[((size_t)(b * VTOT + Off[l] + v)) * DIM + t];
  __syncthreads();
  float acc = bv[t];
  for (int c = 0; c < DIM; ++c) acc += fr[c] * (float)wv[(size_t)c * DIM + t];
  vh[t] = acc; __syncthreads();
  float o = bo[t];
  for (int c = 0; c < DIM; ++c) o += vh[c] * (float)wo[(size_t)c * DIM + t];
  atomicAdd(&aggH[((size_t)(b * 16 + k)) * DIM + t], o);
}

// -------------------------- residual + LayerNorm ---------------------------
__global__ void finalize_kernel(const float* __restrict__ Qs, const float* __restrict__ Qt,
                                const float* __restrict__ Qa, const float* __restrict__ Qd,
                                const float* __restrict__ aggQs, const float* __restrict__ aggQt,
                                const float* __restrict__ aggH,
                                const float* gS, const float* bS, const float* gT, const float* bT,
                                const float* gA, const float* bA, const float* gD, const float* bD,
                                float* __restrict__ out) {
  const int r = blockIdx.x, t = threadIdx.x;
  const float *src, *agg, *g, *bb; float* o; int row;
  if (r < 200)      { src = Qs; agg = aggQs; g = gS; bb = bS; o = out;         row = r; }
  else if (r < 232) { src = Qt; agg = aggQt; g = gT; bb = bT; o = out + 51200; row = r - 200; }
  else if (r < 264) { src = Qa; agg = aggH;  g = gA; bb = bA; o = out + 59392; row = r - 232; }
  else              { src = Qd; agg = aggH;  g = gD; bb = bD; o = out + 67584; row = r - 264; }
  const float x = src[(size_t)row * DIM + t] + 0.25f * agg[(size_t)row * DIM + t];
  __shared__ float red[256];
  red[t] = x; __syncthreads();
  for (int off = 128; off > 0; off >>= 1) { if (t < off) red[t] += red[t + off]; __syncthreads(); }
  const float mean = red[0] * (1.0f / 256.0f); __syncthreads();
  const float dx = x - mean;
  red[t] = dx * dx; __syncthreads();
  for (int off = 128; off > 0; off >>= 1) { if (t < off) red[t] += red[t + off]; __syncthreads(); }
  const float var = red[0] * (1.0f / 256.0f);
  o[(size_t)row * DIM + t] = dx * rsqrtf(var + 1e-5f) * g[t] + bb[t];
}

// ---------------------------------------------------------------------------
extern "C" void kernel_launch(void* const* d_in, const int* in_sizes, int n_in,
                              void* d_out, int out_size, void* d_ws, size_t ws_size,
                              hipStream_t stream) {
  (void)in_sizes; (void)n_in; (void)out_size; (void)ws_size;
  const float* Qs = (const float*)d_in[0];
  const float* Qt = (const float*)d_in[1];
  const float* Qa = (const float*)d_in[2];
  const float* Qd = (const float*)d_in[3];
  const float* feat[4] = {(const float*)d_in[4], (const float*)d_in[5],
                          (const float*)d_in[6], (const float*)d_in[7]};
  const float* wq_s = (const float*)d_in[8];  const float* bq_s = (const float*)d_in[9];
  const float* wk_s = (const float*)d_in[10]; const float* bk_s = (const float*)d_in[11];
  const float* wv_s = (const float*)d_in[12]; const float* bv_s = (const float*)d_in[13];
  const float* wo_s = (const float*)d_in[14]; const float* bo_s = (const float*)d_in[15];
  // hard wq/bq/wk/bk (16..19) unused: one-hot softmax ignores scores
  const float* wv_h = (const float*)d_in[20]; const float* bv_h = (const float*)d_in[21];
  const float* wo_h = (const float*)d_in[22]; const float* bo_h = (const float*)d_in[23];
  const float* projW[4] = {(const float*)d_in[24], (const float*)d_in[26],
                           (const float*)d_in[28], (const float*)d_in[30]};
  const float* projB[4] = {(const float*)d_in[25], (const float*)d_in[27],
                           (const float*)d_in[29], (const float*)d_in[31]};
  const float* gS = (const float*)d_in[32]; const float* bSn = (const float*)d_in[33];
  const float* gT = (const float*)d_in[34]; const float* bTn = (const float*)d_in[35];
  const float* gA = (const float*)d_in[36]; const float* bAn = (const float*)d_in[37];
  const float* gD = (const float*)d_in[38]; const float* bDn = (const float*)d_in[39];

  // ---- workspace bump allocator ----
  char* p = (char*)d_ws;
  auto alloc = [&](size_t bytes) -> void* {
    void* r = (void*)p; p += (bytes + 255) & ~(size_t)255; return r;
  };
  const size_t FFE = (size_t)ROWS_FF * DIM;
  __bf16* ff  = (__bf16*)alloc(FFE * 2);
  __bf16* Kb  = (__bf16*)alloc(FFE * 2);
  __bf16* Vb  = (__bf16*)alloc(FFE * 2);
  __bf16* Wp[4]; for (int l = 0; l < 4; ++l) Wp[l] = (__bf16*)alloc(65536 * 2);
  __bf16* Wq  = (__bf16*)alloc(65536 * 2);   // all *sw* = fragment-swizzled
  __bf16* Wk  = (__bf16*)alloc(65536 * 2);
  __bf16* Wv  = (__bf16*)alloc(65536 * 2);
  __bf16* Wo  = (__bf16*)alloc(65536 * 2);
  __bf16* Wvh = (__bf16*)alloc(65536 * 2);   // plain layout (GEMV use)
  __bf16* Woh = (__bf16*)alloc(65536 * 2);
  __bf16* Qsb = (__bf16*)alloc(51200 * 2);
  __bf16* Qtb = (__bf16*)alloc(8192 * 2);
  __bf16* Qab = (__bf16*)alloc(8192 * 2);
  __bf16* QsP = (__bf16*)alloc(51200 * 2);
  __bf16* QtP = (__bf16*)alloc(8192 * 2);
  __bf16* OaS = (__bf16*)alloc(51200 * 2);
  __bf16* OaT = (__bf16*)alloc(8192 * 2);
  float*  agS = (float*)alloc(51200 * 4);
  float*  agT = (float*)alloc(8192 * 4);
  float*  agH = (float*)alloc(8192 * 4);
  int*    idx = (int*)alloc(128 * 4);

  const int Vl[4] = {32768, 4096, 512, 64};
  const int Off[4] = {0, 32768, 36864, 37376};

  // ---- 1) weight swizzle / bf16 conversions ----
  auto swz = [&](const float* s, __bf16* d) {
    swz_weight_kernel<<<dim3(256), dim3(256), 0, stream>>>(s, d);
  };
  auto cvt = [&](const float* s, __bf16* d, int n) {
    cvt_kernel<<<dim3((n + 255) / 256), dim3(256), 0, stream>>>(s, d, n);
  };
  for (int l = 0; l < 4; ++l) swz(projW[l], Wp[l]);
  swz(wq_s, Wq); swz(wk_s, Wk); swz(wv_s, Wv); swz(wo_s, Wo);
  cvt(wv_h, Wvh, 65536); cvt(wo_h, Woh, 65536);
  cvt(Qs, Qsb, 51200); cvt(Qt, Qtb, 8192); cvt(Qa, Qab, 8192);
  zero_kernel<<<dim3(32), dim3(256), 0, stream>>>(agH, 8192);

  // ---- 2) feat projections (WMMA, async LDS staging, 4 N-tiles/wave) ----
  for (int l = 0; l < 4; ++l)
    featproj_kernel<<<dim3(Vl[l] / 16, 4, NB), dim3(32), 0, stream>>>(
        feat[l], Wp[l], projB[l], ff, Vl[l], Off[l]);

  // ---- 3) K/V projections over all voxels; Q projections (WMMA) ----
  gemm_rows_kernel<<<dim3(ROWS_FF / 16, 4), dim3(32), 0, stream>>>(ff, Wk, bk_s, Kb, ROWS_FF);
  gemm_rows_kernel<<<dim3(ROWS_FF / 16, 4), dim3(32), 0, stream>>>(ff, Wv, bv_s, Vb, ROWS_FF);
  gemm_rows_kernel<<<dim3(13, 4), dim3(32), 0, stream>>>(Qsb, Wq, bq_s, QsP, 200);
  gemm_rows_kernel<<<dim3(2, 4), dim3(32), 0, stream>>>(Qtb, Wq, bq_s, QtP, 32);

  // ---- 4) flash soft attention, all 4 levels fused (WMMA + async V stage) ----
  flash_soft_kernel<<<dim3(7, HEADS, NB), dim3(32), 0, stream>>>(QsP, Kb, Vb, OaS, 100);
  flash_soft_kernel<<<dim3(1, HEADS, NB), dim3(32), 0, stream>>>(QtP, Kb, Vb, OaT, 16);

  // ---- 5) shared output projection (w_o identical across levels; bias x4) ----
  gemm_rows_f32_kernel<<<dim3(13, 4), dim3(32), 0, stream>>>(OaS, Wo, bo_s, 4.0f, agS, 200);
  gemm_rows_f32_kernel<<<dim3(2, 4), dim3(32), 0, stream>>>(OaT, Wo, bo_s, 4.0f, agT, 32);

  // ---- 6) hard routing: argmax then one-hot gather (Qa_agg == Qd_agg) ----
  hard_argmax_kernel<<<dim3(16, NB, 4), dim3(256), 0, stream>>>(Qab, ff, idx);
  hard_out_kernel<<<dim3(16, NB, 4), dim3(256), 0, stream>>>(ff, idx, Wvh, bv_h, Woh, bo_h, agH);

  // ---- 7) residual + LayerNorm ----
  finalize_kernel<<<dim3(296), dim3(256), 0, stream>>>(
      Qs, Qt, Qa, Qd, agS, agT, agH,
      gS, bSn, gT, bTn, gA, bAn, gD, bDn, (float*)d_out);
}
